// GraphAutoencoder_14156212208321
// MI455X (gfx1250) — matrix operand order repaired and checked by reading, not compiled, not run on hardware
//
#include <hip/hip_runtime.h>
#include <hip/hip_bf16.h>

// ---------------------------------------------------------------------------
// MI455X (gfx1250) implementation.
// Precision plan: activations bf16, accumulation f32 via v_wmma_f32_16x16x32_bf16.
// Memory-bound workload -> bf16 halves activation traffic; weights (<2MB bf16)
// are transposed once per launch into [N,K] so WMMA A/B fragments are
// contiguous 16B loads in contiguous k order (k = 16h + e, same for A and B; wave32).
// N-blocked GEMM: each wave computes a 16x64 strip (4 accumulators), reusing
// the A fragment across 4 WMMAs -> 4x less activation traffic.
// ---------------------------------------------------------------------------

typedef __attribute__((ext_vector_type(16))) __bf16 v16bf;
typedef __attribute__((ext_vector_type(8)))  __bf16 v8bf;
typedef __attribute__((ext_vector_type(8)))  float  v8f;

#define NNODE0 65536
#define NTOT   131072
#define NEDGE  1048576

// ---------------- elementwise / utility kernels ----------------------------

__global__ void fill_f32_kernel(float* __restrict__ p, float v, long long n) {
  long long i = (long long)blockIdx.x * blockDim.x + threadIdx.x;
  long long stride = (long long)gridDim.x * blockDim.x;
  for (; i < n; i += stride) p[i] = v;
}

__global__ void cvt_f32_to_bf16_kernel(const float* __restrict__ src,
                                       __bf16* __restrict__ dst, long long n) {
  long long i = (long long)blockIdx.x * blockDim.x + threadIdx.x;
  long long stride = (long long)gridDim.x * blockDim.x;
  for (; i < n; i += stride) dst[i] = (__bf16)src[i];
}

// f32 src [K,N] (row-major) -> bf16 dst [rows,K], column n of src lands in
// dst row (n + rowOff); rows outside [rowOff, rowOff+N) are zero.
__global__ void transpose_to_bf16_kernel(const float* __restrict__ src,
                                         __bf16* __restrict__ dst,
                                         int K, int N, int rowOff, int rows) {
  int idx = blockIdx.x * blockDim.x + threadIdx.x;
  int total = rows * K;
  if (idx >= total) return;
  int n = idx / K;
  int k = idx - n * K;
  float v = 0.f;
  int sn = n - rowOff;
  if (sn >= 0 && sn < N) v = src[(size_t)k * N + sn];
  dst[idx] = (__bf16)v;
}

// bias for field encoder: [zeros(8) | bf(120)] per type
__global__ void pad_bias_kernel(const float* __restrict__ bf, float* __restrict__ dst) {
  int t = blockIdx.x;
  int i = threadIdx.x;   // 128 threads
  dst[t * 128 + i] = (i < 8) ? 0.f : bf[t * 120 + (i - 8)];
}

__global__ void degree_kernel(const int* __restrict__ edst, float* __restrict__ deg, int E) {
  int i = blockIdx.x * blockDim.x + threadIdx.x;
  if (i < E) atomicAdd(&deg[edst[i]], 1.0f);
}

// agg[dst[e], c] += bn[src[e], c]  (64 columns/edge, coalesced per half-wave)
__global__ void scatter_bn_kernel(const __bf16* __restrict__ bn,
                                  const int* __restrict__ esrc,
                                  const int* __restrict__ edst,
                                  float* __restrict__ agg, int E) {
  long long idx = (long long)blockIdx.x * blockDim.x + threadIdx.x;
  if (idx >= (long long)E * 64) return;
  int e = (int)(idx >> 6);
  int c = (int)(idx & 63);
  float v = (float)bn[(size_t)esrc[e] * 64 + c];
  atomicAdd(&agg[(size_t)edst[e] * 64 + c], v);
}

__global__ void normalize_kernel(const float* __restrict__ agg,
                                 const float* __restrict__ deg,
                                 __bf16* __restrict__ out, long long total) {
  long long idx = (long long)blockIdx.x * blockDim.x + threadIdx.x;
  if (idx >= total) return;
  float d = deg[idx >> 6];
  if (d < 1.f) d = 1.f;
  out[idx] = (__bf16)(agg[idx] / d);
}

// ---------------- WMMA GEMM --------------------------------------------------
// C[M,N] = act(A[M,K](bf16, ldA) @ W[K,N] + bias), W given transposed as
// Wt[N,K] bf16 row-major. One wave computes a 16x64 strip (4 x 16x16 tiles,
// A fragment reused across the 4 WMMAs). K-loop step 32 via
// v_wmma_f32_16x16x32_bf16 (f32 accum). Fragment layouts (contiguous k order used by this code):
//   A: lane(h=lane>>4, m=lane&15): elem e    -> A[m][k0+16h+e]
//   B: lane(h, n=lane&15):        elem e    -> W[k0+16h+e][n] = Wt[n][k0+16h+e]
//   C: vgpr r, lane(h,n):         C[r+8h][n]
__global__ __launch_bounds__(256)
void wmma_gemm_bias_act_kernel(const __bf16* __restrict__ A, int ldA,
                               const __bf16* __restrict__ Wt,   // [N,K]
                               const float*  __restrict__ bias, // [N]
                               __bf16* __restrict__ Cb, float* __restrict__ Cf,
                               int ldC, int M, int N, int K, int do_relu) {
  const int lane = threadIdx.x & 31;
  const int wave = threadIdx.x >> 5;
  const int nstrip = N >> 6;                  // 64-wide strips
  const int wl = blockIdx.x * 8 + wave;
  const int tm = wl / nstrip;
  const int ts = wl - tm * nstrip;            // strip index
  if (tm * 16 >= M) return;

  const int h = lane >> 4;   // lane half selects K sub-block
  const int l = lane & 15;   // A: row within tile / B,C: column within tile

  const __bf16* __restrict__ Arow = A + (size_t)(tm * 16 + l) * ldA;
  const __bf16* __restrict__ Bcol[4];
#pragma unroll
  for (int j = 0; j < 4; ++j)
    Bcol[j] = Wt + (size_t)(ts * 64 + j * 16 + l) * K;

  v8f acc0 = {}, acc1 = {}, acc2 = {}, acc3 = {};
  for (int k0 = 0; k0 < K; k0 += 32) {
    __builtin_prefetch(Arow + k0 + 64, 0, 1);   // global_prefetch_b8 (next chunk)
    v8bf alo = *(const v8bf*)(Arow + k0 + 16 * h);
    v8bf ahi = *(const v8bf*)(Arow + k0 + 16 * h + 8);
    v16bf a;
#pragma unroll
    for (int i = 0; i < 8; ++i) { a[i] = alo[i]; a[8 + i] = ahi[i]; }

    v16bf b[4];
#pragma unroll
    for (int j = 0; j < 4; ++j) {
      v8bf blo = *(const v8bf*)(Bcol[j] + k0 + 16 * h);
      v8bf bhi = *(const v8bf*)(Bcol[j] + k0 + 16 * h + 8);
#pragma unroll
      for (int i = 0; i < 8; ++i) { b[j][i] = blo[i]; b[j][8 + i] = bhi[i]; }
    }
    acc0 = __builtin_amdgcn_wmma_f32_16x16x32_bf16(false, a, false, b[0], (short)0, acc0, false, false);
    acc1 = __builtin_amdgcn_wmma_f32_16x16x32_bf16(false, a, false, b[1], (short)0, acc1, false, false);
    acc2 = __builtin_amdgcn_wmma_f32_16x16x32_bf16(false, a, false, b[2], (short)0, acc2, false, false);
    acc3 = __builtin_amdgcn_wmma_f32_16x16x32_bf16(false, a, false, b[3], (short)0, acc3, false, false);
  }

  v8f accs[4] = {acc0, acc1, acc2, acc3};
#pragma unroll
  for (int j = 0; j < 4; ++j) {
    const int col = ts * 64 + j * 16 + l;
    const float bv = bias ? bias[col] : 0.f;
#pragma unroll
    for (int r = 0; r < 8; ++r) {
      float v = accs[j][r] + bv;
      if (do_relu) v = v > 0.f ? v : 0.f;
      const size_t row = (size_t)(tm * 16 + r + 8 * h);
      if (Cb) Cb[row * ldC + col] = (__bf16)v;
      if (Cf) Cf[row * ldC + col] = v;
    }
  }
}

// ---------------- host orchestration ----------------------------------------

extern "C" void kernel_launch(void* const* d_in, const int* in_sizes, int n_in,
                              void* d_out, int out_size, void* d_ws, size_t ws_size,
                              hipStream_t stream) {
  const float* feat0 = (const float*)d_in[0];
  const float* feat1 = (const float*)d_in[1];
  const float* Wf    = (const float*)d_in[2];
  const float* bfv   = (const float*)d_in[3];
  const float* A0We1 = (const float*)d_in[4];
  const float* A0be1 = (const float*)d_in[5];
  const float* A0We2 = (const float*)d_in[6];
  const float* A0be2 = (const float*)d_in[7];
  const float* A0Wd1 = (const float*)d_in[8];
  const float* A0bd1 = (const float*)d_in[9];
  const float* A0Wd2 = (const float*)d_in[10];
  const float* A0bd2 = (const float*)d_in[11];
  const float* AWe1  = (const float*)d_in[12];
  const float* Abe1  = (const float*)d_in[13];
  const float* AWe2  = (const float*)d_in[14];
  const float* Abe2  = (const float*)d_in[15];
  const float* AWd1  = (const float*)d_in[16];
  const float* Abd1  = (const float*)d_in[17];
  const float* AWd2  = (const float*)d_in[18];
  const float* Abd2  = (const float*)d_in[19];
  const float* Ws    = (const float*)d_in[20];
  const float* bs    = (const float*)d_in[21];
  const float* Wp    = (const float*)d_in[22];
  const float* bp    = (const float*)d_in[23];
  const float* Wdec  = (const float*)d_in[24];
  const float* bdec  = (const float*)d_in[25];
  const int*   esrc  = (const int*)d_in[26];
  const int*   edst  = (const int*)d_in[27];
  float* out = (float*)d_out;

  // ---- workspace carve ----
  char* ws = (char*)d_ws;
  size_t off = 0;
  auto carve = [&](size_t bytes) -> char* {
    char* p = ws + off;
    off += (bytes + 255) & ~(size_t)255;
    return p;
  };

  // transposed bf16 weights
  __bf16* Wft[2];     for (int t = 0; t < 2; ++t) Wft[t]    = (__bf16*)carve((size_t)128 * 256 * 2);
  __bf16* A0We1t[2];  for (int t = 0; t < 2; ++t) A0We1t[t] = (__bf16*)carve((size_t)256 * 128 * 2);
  __bf16* A0We2t[2];  for (int t = 0; t < 2; ++t) A0We2t[t] = (__bf16*)carve((size_t)64  * 256 * 2);
  __bf16* A0Wd1t[2];  for (int t = 0; t < 2; ++t) A0Wd1t[t] = (__bf16*)carve((size_t)256 * 64  * 2);
  __bf16* A0Wd2t[2];  for (int t = 0; t < 2; ++t) A0Wd2t[t] = (__bf16*)carve((size_t)128 * 256 * 2);
  __bf16* AWe1t[4];   for (int i = 0; i < 4; ++i) AWe1t[i]  = (__bf16*)carve((size_t)256 * 192 * 2);
  __bf16* AWe2t[4];   for (int i = 0; i < 4; ++i) AWe2t[i]  = (__bf16*)carve((size_t)64  * 256 * 2);
  __bf16* AWd1t[4];   for (int i = 0; i < 4; ++i) AWd1t[i]  = (__bf16*)carve((size_t)256 * 64  * 2);
  __bf16* AWd2t[4];   for (int i = 0; i < 4; ++i) AWd2t[i]  = (__bf16*)carve((size_t)192 * 256 * 2);
  __bf16* Wst  = (__bf16*)carve((size_t)64 * 64 * 2);
  __bf16* Wpt[2];     for (int t = 0; t < 2; ++t) Wpt[t]    = (__bf16*)carve((size_t)128 * 192 * 2);
  __bf16* Wdect[2];   for (int t = 0; t < 2; ++t) Wdect[t]  = (__bf16*)carve((size_t)256 * 128 * 2);
  float*  bias_pad = (float*)carve(2 * 128 * sizeof(float));

  // activations
  __bf16* xbuf[2];    for (int t = 0; t < 2; ++t) xbuf[t] = (__bf16*)carve((size_t)NNODE0 * 192 * 2);
  __bf16* hbuf  = (__bf16*)carve((size_t)NNODE0 * 256 * 2);   // h / g / p / featbf scratch
  __bf16* bnbuf = (__bf16*)carve((size_t)NTOT * 64 * 2);      // bottlenecks (both types)
  float*  agg   = (float*) carve((size_t)NTOT * 64 * 4);
  __bf16* aggn  = (__bf16*)carve((size_t)NTOT * 64 * 2);
  float*  deg   = (float*) carve((size_t)NTOT * 4);

  // ---- launch helpers ----
  auto tr = [&](const float* src, __bf16* dst, int K, int N, int rowOff, int rows) {
    int total = rows * K;
    transpose_to_bf16_kernel<<<(total + 255) / 256, 256, 0, stream>>>(src, dst, K, N, rowOff, rows);
  };
  auto gemm = [&](const __bf16* A, int ldA, const __bf16* Wt, const float* bias,
                  __bf16* Cb, float* Cf, int ldC, int M, int N, int K, int relu) {
    int blocks = (M / 16) * (N / 64) / 8;   // M/16 is always a multiple of 8 here
    wmma_gemm_bias_act_kernel<<<blocks, 256, 0, stream>>>(A, ldA, Wt, bias, Cb, Cf, ldC, M, N, K, relu);
  };

  // ---- 1) convert/transpose weights (resident in L2 afterwards) ----
  for (int t = 0; t < 2; ++t) {
    tr(Wf    + (size_t)t * 256 * 120, Wft[t],    256, 120, 8, 128);  // zero-pad first 8 rows
    tr(A0We1 + (size_t)t * 128 * 256, A0We1t[t], 128, 256, 0, 256);
    tr(A0We2 + (size_t)t * 256 * 64,  A0We2t[t], 256, 64,  0, 64);
    tr(A0Wd1 + (size_t)t * 64  * 256, A0Wd1t[t], 64,  256, 0, 256);
    tr(A0Wd2 + (size_t)t * 256 * 128, A0Wd2t[t], 256, 128, 0, 128);
    for (int d = 0; d < 2; ++d) {
      int i = t * 2 + d;
      tr(AWe1 + (size_t)i * 192 * 256, AWe1t[i], 192, 256, 0, 256);
      tr(AWe2 + (size_t)i * 256 * 64,  AWe2t[i], 256, 64,  0, 64);
      tr(AWd1 + (size_t)i * 64  * 256, AWd1t[i], 64,  256, 0, 256);
      tr(AWd2 + (size_t)i * 256 * 192, AWd2t[i], 256, 192, 0, 192);
    }
    tr(Wp   + (size_t)t * 192 * 128, Wpt[t],   192, 128, 0, 128);
    tr(Wdec + (size_t)t * 128 * 256, Wdect[t], 128, 256, 0, 256);
  }
  tr(Ws, Wst, 64, 64, 0, 64);
  pad_bias_kernel<<<2, 128, 0, stream>>>(bfv, bias_pad);

  // ---- 2) depth-0: field encode -> AE0 per type ----
  for (int t = 0; t < 2; ++t) {
    const float* feat = t == 0 ? feat0 : feat1;
    // featbf (reuse hbuf)
    cvt_f32_to_bf16_kernel<<<8192, 256, 0, stream>>>(feat, hbuf, (long long)NNODE0 * 256);
    // x = [0_8 | relu(feat@Wf+bf)] -> xbuf[:, 0:128]     (padded Wft/bias do the concat)
    gemm(hbuf, 256, Wft[t], bias_pad + t * 128, xbuf[t], nullptr, 192, NNODE0, 128, 256, 1);
    // h = relu(x@We1+be1)
    gemm(xbuf[t], 192, A0We1t[t], A0be1 + t * 256, hbuf, nullptr, 256, NNODE0, 256, 128, 1);
    // bn = h@We2+be2
    gemm(hbuf, 256, A0We2t[t], A0be2 + t * 64, bnbuf + (size_t)t * NNODE0 * 64, nullptr, 64, NNODE0, 64, 256, 0);
    // g = relu(bn@Wd1+bd1)
    gemm(bnbuf + (size_t)t * NNODE0 * 64, 64, A0Wd1t[t], A0bd1 + t * 256, hbuf, nullptr, 256, NNODE0, 256, 64, 1);
    // o = g@Wd2+bd2 -> xbuf[:, 0:128]
    gemm(hbuf, 256, A0Wd2t[t], A0bd2 + t * 128, xbuf[t], nullptr, 192, NNODE0, 128, 256, 0);
  }

  // ---- 3) degrees (once) ----
  fill_f32_kernel<<<1024, 256, 0, stream>>>(deg, 0.f, NTOT);
  degree_kernel<<<(NEDGE + 255) / 256, 256, 0, stream>>>(edst, deg, NEDGE);

  auto summarize = [&]() {
    fill_f32_kernel<<<4096, 256, 0, stream>>>(agg, 0.f, (long long)NTOT * 64);
    scatter_bn_kernel<<<(int)(((long long)NEDGE * 64) / 256), 256, 0, stream>>>(bnbuf, esrc, edst, agg, NEDGE);
    normalize_kernel<<<(int)(((long long)NTOT * 64 + 255) / 256), 256, 0, stream>>>(agg, deg, aggn, (long long)NTOT * 64);
    // summ = relu(aggn@Ws+bs) -> xbuf[:, 128:192]
    for (int t = 0; t < 2; ++t)
      gemm(aggn + (size_t)t * NNODE0 * 64, 64, Wst, bs, xbuf[t] + 128, nullptr, 192, NNODE0, 64, 64, 1);
  };

  // ---- 4) depth loop ----
  for (int d = 0; d < 2; ++d) {
    summarize();   // uses previous-iteration bottlenecks
    for (int t = 0; t < 2; ++t) {
      int i = t * 2 + d;
      // h = relu(x@We1+be1), K=192
      gemm(xbuf[t], 192, AWe1t[i], Abe1 + (size_t)i * 256, hbuf, nullptr, 256, NNODE0, 256, 192, 1);
      // bn = h@We2+be2
      gemm(hbuf, 256, AWe2t[i], Abe2 + (size_t)i * 64, bnbuf + (size_t)t * NNODE0 * 64, nullptr, 64, NNODE0, 64, 256, 0);
      // g = relu(bn@Wd1+bd1)
      gemm(bnbuf + (size_t)t * NNODE0 * 64, 64, AWd1t[i], Abd1 + (size_t)i * 256, hbuf, nullptr, 256, NNODE0, 256, 64, 1);
      // o = g@Wd2+bd2 -> xbuf[:, 0:192] (cols 128+ later replaced by summ)
      gemm(hbuf, 256, AWd2t[i], Abd2 + (size_t)i * 192, xbuf[t], nullptr, 192, NNODE0, 192, 256, 0);
    }
  }

  // ---- 5) final: summarize, project, decode ----
  summarize();
  for (int t = 0; t < 2; ++t) {
    // p = relu(x@Wp+bp) -> hbuf [65536,128]
    gemm(xbuf[t], 192, Wpt[t], bp + t * 128, hbuf, nullptr, 128, NNODE0, 128, 192, 1);
    // recon = p@Wdec+bdec -> d_out (f32)
    gemm(hbuf, 128, Wdect[t], bdec + t * 256, nullptr, out + (size_t)t * NNODE0 * 256, 256, NNODE0, 256, 128, 0);
  }
}
